// LSTM_FEAT_24524263260272
// MI455X (gfx1250) — matrix operand order repaired and checked by reading, not compiled, not run on hardware
//
#include <hip/hip_runtime.h>

typedef __attribute__((ext_vector_type(16))) __bf16 v16bf;
typedef __attribute__((ext_vector_type(8)))  __bf16 v8bf;
typedef __attribute__((ext_vector_type(8)))  float  v8f;

#define T_STEPS 4096
#define IN_DIM  360
#define KPAD    384
#define G4H     8192      // 4*HIDDEN
#define HID     2048
#define OUTD    512
#define EMB     64
#define LIN0    424
#define NB_SCAN 64
#define NT_SCAN 256

// ---------------- workspace layout (bytes) ----------------
#define OFF_GX   ((size_t)0)                                   // T*G4H*4   = 134 MB
#define OFF_INB  (OFF_GX  + (size_t)T_STEPS * G4H  * 4)        // T*KPAD*2  = 3 MB
#define OFF_WXB  (OFF_INB + (size_t)T_STEPS * KPAD * 2)        // G4H*KPAD*2= 6 MB
#define OFF_WEF  (OFF_WXB + (size_t)G4H * KPAD * 2)            // G4H*EMB*4 = 2 MB
#define OFF_W1B  (OFF_WEF + (size_t)G4H * EMB * 4)             // G4H*HID*2 = 33.5 MB
#define OFF_WLB  (OFF_W1B + (size_t)G4H * HID * 2)             // OUTD*HID*2= 2 MB
#define OFF_WMB  (OFF_WLB + (size_t)OUTD * HID * 2)            // EMB*OUTD*2
#define OFF_B0   (OFF_WMB + (size_t)EMB * OUTD * 2)            // G4H*4
#define OFF_B1   (OFF_B0  + (size_t)G4H * 4)                   // G4H*4
#define OFF_H0   (OFF_B1  + (size_t)G4H * 4)                   // HID*4
#define OFF_H1   (OFF_H0  + (size_t)HID * 4)                   // HID*4
#define OFF_SYNC (OFF_H1  + (size_t)HID * 4)                   // 512 B

__device__ __forceinline__ float sigmoidf_(float x) {
    return 1.0f / (1.0f + __expf(-x));
}

// ---- CDNA5 async global->LDS copy (ASYNCcnt-tracked, per-lane gather) ----
// lds_off: LDS byte offset (low 32 bits of generic LDS address), gptr: global src.
__device__ __forceinline__ void async_g2l_b128(unsigned lds_off, const void* gptr) {
    unsigned long long ga = (unsigned long long)(uintptr_t)gptr;
    asm volatile("global_load_async_to_lds_b128 %0, %1, off"
                 :: "v"(lds_off), "v"(ga) : "memory");
}
__device__ __forceinline__ void wait_async0() {
    asm volatile("s_wait_asynccnt 0x0" ::: "memory");
}
__device__ __forceinline__ unsigned lds_off32(const void* p) {
    return (unsigned)(uintptr_t)p;   // generic LDS addr: low 32 bits = LDS offset
}

// ---------------- prep kernels ----------------
__global__ void k_prep_misc(const float* __restrict__ bih0, const float* __restrict__ bhh0,
                            const float* __restrict__ bih1, const float* __restrict__ bhh1,
                            float* __restrict__ b0, float* __restrict__ b1,
                            unsigned* __restrict__ sync) {
    int i = blockIdx.x * blockDim.x + threadIdx.x;
    if (i < G4H) {
        b0[i] = bih0[i] + bhh0[i];
        b1[i] = bih1[i] + bhh1[i];
    }
    if (i < 128) sync[i] = 0u;   // barrier count + generation
}

// dst[r*dstld + c] = (c < used) ? bf16(src[r*srcld + c]) : 0
__global__ void k_cvt_pad(__bf16* __restrict__ dst, const float* __restrict__ src,
                          int rows, int used, int srcld, int dstld) {
    size_t idx = (size_t)blockIdx.x * blockDim.x + threadIdx.x;
    size_t n = (size_t)rows * dstld;
    if (idx >= n) return;
    int r = (int)(idx / dstld), c = (int)(idx % dstld);
    dst[idx] = (c < used) ? (__bf16)src[(size_t)r * srcld + c] : (__bf16)0.0f;
}

__global__ void k_cvt_flat(__bf16* __restrict__ dst, const float* __restrict__ src, size_t n) {
    size_t idx = (size_t)blockIdx.x * blockDim.x + threadIdx.x;
    if (idx < n) dst[idx] = (__bf16)src[idx];
}

// We (embed columns of Wih0) kept fp32: we[j*EMB+e] = Wih0[j*LIN0 + IN_DIM + e]
__global__ void k_cvt_we(float* __restrict__ we, const float* __restrict__ wih0) {
    size_t idx = (size_t)blockIdx.x * blockDim.x + threadIdx.x;
    if (idx >= (size_t)G4H * EMB) return;
    int j = (int)(idx / EMB), e = (int)(idx % EMB);
    we[idx] = wih0[(size_t)j * LIN0 + IN_DIM + e];
}

// ---------------- gates_x GEMM (WMMA bf16, LDS-staged A) ----------------
// C[m, n] = sum_k A[m,k] * Wx[n,k] + b0[n], over n in [0,2048) u [4096,8192)
// Block = 8 waves sharing one 16-row A strip: stage A (16 x KPAD bf16 = 12 KB)
// into LDS once via async copy, then each wave reads fragments from LDS while
// streaming its private B columns from global (K-contiguous 32 B per lane).
__global__ __launch_bounds__(256) void k_gatesx(const __bf16* __restrict__ Abf,
                                                const __bf16* __restrict__ Wxb,
                                                const float* __restrict__ b0,
                                                float* __restrict__ gx) {
    __shared__ __bf16 sA[16 * KPAD];               // 12 KB

    const int tid = threadIdx.x;
    const int lane = tid & 31;
    const int wave = tid >> 5;
    const int m0 = blockIdx.x * 16;
    const int sy = blockIdx.y;                     // 0..47 -> skip unused f-gate cols
    const int colbase = (sy < 16) ? sy * 128 : sy * 128 + 2048;
    const int n0 = colbase + wave * 16;

    // Async-stage A strip: 16*KPAD*2 bytes = 768 x 16B chunks, 3 per thread.
    {
        const __bf16* abase = Abf + (size_t)m0 * KPAD;
#pragma unroll
        for (int it = 0; it < 3; ++it) {
            int i = tid + it * 256;                // chunk id 0..767
            int elem = i * 8;                      // 8 bf16 per 16B chunk
            async_g2l_b128(lds_off32(&sA[elem]), abase + elem);
        }
    }
    wait_async0();
    __syncthreads();

    // A fragment (16x32 bf16): lanes 0-15 K base 0, lanes 16-31 K base 16
    const int arow = lane & 15;
    const int asel = (lane >> 4) << 4;             // 0 or 16
    // B fragment (32x16): lane = column, lanes 0-15 K 0..15, lanes 16-31 K 16..31
    const int bcol = n0 + (lane & 15);
    const int bsel = (lane >> 4) << 4;             // 0 or 16

    const __bf16* aptr = sA + arow * KPAD + asel;
    const __bf16* bptr = Wxb + (size_t)bcol * KPAD + bsel;

    v8f c = {};
#pragma unroll
    for (int k0 = 0; k0 < KPAD; k0 += 32) {
        v8bf alo = *(const v8bf*)(aptr + k0);          // K = k0+asel .. +7   (LDS)
        v8bf ahi = *(const v8bf*)(aptr + k0 + 8);      // K = k0+8+asel .. +7 (LDS)
        v16bf a = __builtin_shufflevector(alo, ahi, 0, 1, 2, 3, 4, 5, 6, 7,
                                          8, 9, 10, 11, 12, 13, 14, 15);
        v16bf b = *(const v16bf*)(bptr + k0);          // K = k0+bsel .. +15 (global)
        c = __builtin_amdgcn_wmma_f32_16x16x32_bf16(false, a, false, b,
                                                    (short)0, c, false, false);
    }

    const float bias = b0[bcol];
    const int moff = (lane >> 4) << 3;             // lanes 16-31 hold rows M=8..15
#pragma unroll
    for (int r = 0; r < 8; ++r) {
        gx[(size_t)(m0 + moff + r) * G4H + bcol] = c[r] + bias;
    }
}

// ---------------- grid-wide barrier ----------------
__device__ __forceinline__ void grid_sync(unsigned* cnt, unsigned* gen, unsigned nb) {
    __syncthreads();
    if (threadIdx.x == 0) {
        __threadfence();
        unsigned g = *(volatile unsigned*)gen;
        unsigned arrived = atomicAdd(cnt, 1u);
        if (arrived == nb - 1) {
            *cnt = 0;
            __threadfence();
            atomicAdd(gen, 1u);
        } else {
            while (*(volatile unsigned*)gen == g) { __builtin_amdgcn_s_sleep(2); }
        }
        __threadfence();
    }
    __syncthreads();
}

// ---------------- persistent sequential scan ----------------
__global__ __launch_bounds__(NT_SCAN) void k_scan(
        const float* __restrict__ gx, const float* __restrict__ wef,
        const __bf16* __restrict__ w1, const __bf16* __restrict__ wl,
        const __bf16* __restrict__ wm, const float* __restrict__ b1,
        const float* __restrict__ blin, const float* __restrict__ bmap,
        const float* __restrict__ embed0,
        float* __restrict__ h0, float* __restrict__ h1,
        float* __restrict__ out, unsigned* __restrict__ sync) {
    __shared__ float s_emb[EMB];
    __shared__ float s_p[OUTD];
    __shared__ float s_red[NT_SCAN];
    __shared__ float s_h0[HID] __attribute__((aligned(32)));   // 8 KB, async-staged
    __shared__ float s_h1[HID] __attribute__((aligned(32)));   // 8 KB, async-staged

    const int tid = threadIdx.x;
    const int bid = blockIdx.x;
    const int lane = tid & 31;
    const int wave = tid >> 5;
    const int gwave = bid * 8 + wave;              // 0..511
    unsigned* cnt = sync;
    unsigned* gen = sync + 64;

    for (int t = 0; t < T_STEPS; ++t) {
        // ---- Phase A: embed for step t (recomputed redundantly per block) ----
        if (t == 0) {
            for (int e = tid; e < EMB; e += NT_SCAN) s_emb[e] = embed0[e];
        } else {
            const float* y = out + (size_t)(t - 1) * OUTD;
            float m = -1e30f;
            for (int o = tid; o < OUTD; o += NT_SCAN) m = fmaxf(m, y[o]);
            s_red[tid] = m; __syncthreads();
            for (int s = 128; s > 0; s >>= 1) {
                if (tid < s) s_red[tid] = fmaxf(s_red[tid], s_red[tid + s]);
                __syncthreads();
            }
            float mx = s_red[0]; __syncthreads();
            float sum = 0.f;
            for (int o = tid; o < OUTD; o += NT_SCAN) sum += __expf(y[o] - mx);
            s_red[tid] = sum; __syncthreads();
            for (int s = 128; s > 0; s >>= 1) {
                if (tid < s) s_red[tid] += s_red[tid + s];
                __syncthreads();
            }
            float lse = __logf(s_red[0]) + mx; __syncthreads();
            for (int o = tid; o < OUTD; o += NT_SCAN) s_p[o] = y[o] - lse;
            __syncthreads();
            // embed[e] = bmap[e] + Wmap[e,:] . p     (4 lanes per e)
            int e = tid >> 2, sub = tid & 3;
            const __bf16* row = wm + (size_t)e * OUTD + sub * 128;
            float acc = 0.f;
#pragma unroll 8
            for (int o = 0; o < 128; ++o) acc += (float)row[o] * s_p[sub * 128 + o];
            acc += __shfl_xor(acc, 1, 4);
            acc += __shfl_xor(acc, 2, 4);
            if (sub == 0) s_emb[e] = acc + bmap[e];
        }
        __syncthreads();

        // ---- Phase B: h0[k] = cell(gx[t] + We @ embed), f-gate skipped ----
        {
            int kl = tid >> 3, sub = tid & 7;      // 32 k per block, 8 lanes per k
            int k = bid * 32 + kl;
            const float* gxr = gx + (size_t)t * G4H;
            float di = 0.f, dg = 0.f, dd = 0.f;
            const float* wi = wef + (size_t)k * EMB;
            const float* wg = wef + (size_t)(k + 2 * HID) * EMB;
            const float* wo = wef + (size_t)(k + 3 * HID) * EMB;
#pragma unroll
            for (int j = 0; j < 8; ++j) {
                int e = sub * 8 + j;
                float em = s_emb[e];
                di += wi[e] * em;
                dg += wg[e] * em;
                dd += wo[e] * em;
            }
            for (int m = 4; m >= 1; m >>= 1) {
                di += __shfl_xor(di, m, 8);
                dg += __shfl_xor(dg, m, 8);
                dd += __shfl_xor(dd, m, 8);
            }
            if (sub == 0) {
                float gi = gxr[k] + di;
                float gg = gxr[k + 2 * HID] + dg;
                float go = gxr[k + 3 * HID] + dd;
                float cc = sigmoidf_(gi) * tanhf(gg);
                h0[k] = sigmoidf_(go) * tanhf(cc);
            }
        }
        grid_sync(cnt, gen, NB_SCAN);

        // ---- async-stage full h0 vector into LDS (2 x b128 per thread) ----
#pragma unroll
        for (int it = 0; it < 2; ++it) {
            int i = tid + it * NT_SCAN;            // float4 chunk 0..511
            async_g2l_b128(lds_off32(&s_h0[i * 4]), h0 + i * 4);
        }
        wait_async0();
        __syncthreads();

        // ---- Phase C: h1[k] = cell(Wih1 @ h0 + b1), 3 row-dots per k, 4 k per wave
#pragma unroll
        for (int kk = 0; kk < 4; ++kk) {
            int k = gwave * 4 + kk;
            const __bf16* ri = w1 + (size_t)k * HID;
            const __bf16* rg = w1 + (size_t)(k + 2 * HID) * HID;
            const __bf16* ro = w1 + (size_t)(k + 3 * HID) * HID;
            __builtin_prefetch(ri + HID, 0, 0);    // next k's i-row
            float di = 0.f, dg = 0.f, dd = 0.f;
#pragma unroll
            for (int c8 = 0; c8 < 8; ++c8) {
                int base = c8 * 256 + lane * 8;
                v8bf wvi = *(const v8bf*)(ri + base);
                v8bf wvg = *(const v8bf*)(rg + base);
                v8bf wvo = *(const v8bf*)(ro + base);
                v8f hv = *(const v8f*)(&s_h0[base]);
#pragma unroll
                for (int j = 0; j < 8; ++j) {
                    di += (float)wvi[j] * hv[j];
                    dg += (float)wvg[j] * hv[j];
                    dd += (float)wvo[j] * hv[j];
                }
            }
            for (int m = 16; m >= 1; m >>= 1) {
                di += __shfl_xor(di, m, 32);
                dg += __shfl_xor(dg, m, 32);
                dd += __shfl_xor(dd, m, 32);
            }
            if (lane == 0) {
                float gi = di + b1[k];
                float gg = dg + b1[k + 2 * HID];
                float go = dd + b1[k + 3 * HID];
                h1[k] = sigmoidf_(go) * tanhf(sigmoidf_(gi) * tanhf(gg));
            }
        }
        grid_sync(cnt, gen, NB_SCAN);

        // ---- async-stage full h1 vector into LDS ----
#pragma unroll
        for (int it = 0; it < 2; ++it) {
            int i = tid + it * NT_SCAN;
            async_g2l_b128(lds_off32(&s_h1[i * 4]), h1 + i * 4);
        }
        wait_async0();
        __syncthreads();

        // ---- Phase D: y[o] = Wlin[o,:] . h1 + blin[o], one output per wave ----
        {
            int o = gwave;                          // 512 waves, 512 outputs
            const __bf16* rw = wl + (size_t)o * HID;
            float acc = 0.f;
#pragma unroll
            for (int c8 = 0; c8 < 8; ++c8) {
                int base = c8 * 256 + lane * 8;
                v8bf wv = *(const v8bf*)(rw + base);
                v8f hv = *(const v8f*)(&s_h1[base]);
#pragma unroll
                for (int j = 0; j < 8; ++j) acc += (float)wv[j] * hv[j];
            }
            for (int m = 16; m >= 1; m >>= 1) acc += __shfl_xor(acc, m, 32);
            if (lane == 0) out[(size_t)t * OUTD + o] = acc + blin[o];
        }
        grid_sync(cnt, gen, NB_SCAN);              // y visible for next Phase A
    }
}

// ---------------- host launch ----------------
extern "C" void kernel_launch(void* const* d_in, const int* in_sizes, int n_in,
                              void* d_out, int out_size, void* d_ws, size_t ws_size,
                              hipStream_t stream) {
    const float* inputVecs = (const float*)d_in[0];
    const float* Wih0 = (const float*)d_in[1];
    const float* bih0 = (const float*)d_in[2];
    const float* bhh0 = (const float*)d_in[3];
    const float* Wih1 = (const float*)d_in[4];
    const float* bih1 = (const float*)d_in[5];
    const float* bhh1 = (const float*)d_in[6];
    const float* Wlin = (const float*)d_in[7];
    const float* blin = (const float*)d_in[8];
    const float* Wmap = (const float*)d_in[9];
    const float* bmap = (const float*)d_in[10];
    const float* embed0 = (const float*)d_in[11];
    float* out = (float*)d_out;

    char* ws = (char*)d_ws;
    float*    gx  = (float*)(ws + OFF_GX);
    __bf16*   inb = (__bf16*)(ws + OFF_INB);
    __bf16*   wxb = (__bf16*)(ws + OFF_WXB);
    float*    wef = (float*)(ws + OFF_WEF);
    __bf16*   w1b = (__bf16*)(ws + OFF_W1B);
    __bf16*   wlb = (__bf16*)(ws + OFF_WLB);
    __bf16*   wmb = (__bf16*)(ws + OFF_WMB);
    float*    b0  = (float*)(ws + OFF_B0);
    float*    b1  = (float*)(ws + OFF_B1);
    float*    h0  = (float*)(ws + OFF_H0);
    float*    h1  = (float*)(ws + OFF_H1);
    unsigned* syn = (unsigned*)(ws + OFF_SYNC);

    // prep (all independent, stream-ordered before GEMM/scan)
    k_prep_misc<<<(G4H + 255) / 256, 256, 0, stream>>>(bih0, bhh0, bih1, bhh1, b0, b1, syn);
    {
        size_t n = (size_t)T_STEPS * KPAD;
        k_cvt_pad<<<(unsigned)((n + 255) / 256), 256, 0, stream>>>(inb, inputVecs,
                                                                   T_STEPS, IN_DIM, IN_DIM, KPAD);
    }
    {
        size_t n = (size_t)G4H * KPAD;
        k_cvt_pad<<<(unsigned)((n + 255) / 256), 256, 0, stream>>>(wxb, Wih0,
                                                                   G4H, IN_DIM, LIN0, KPAD);
    }
    {
        size_t n = (size_t)G4H * EMB;
        k_cvt_we<<<(unsigned)((n + 255) / 256), 256, 0, stream>>>(wef, Wih0);
    }
    {
        size_t n = (size_t)G4H * HID;
        k_cvt_flat<<<(unsigned)((n + 255) / 256), 256, 0, stream>>>(w1b, Wih1, n);
    }
    {
        size_t n = (size_t)OUTD * HID;
        k_cvt_flat<<<(unsigned)((n + 255) / 256), 256, 0, stream>>>(wlb, Wlin, n);
    }
    {
        size_t n = (size_t)EMB * OUTD;
        k_cvt_flat<<<(unsigned)((n + 255) / 256), 256, 0, stream>>>(wmb, Wmap, n);
    }

    // gates_x GEMM: M tiles = 4096/16 = 256; N supertiles of 128 cols, f-gates skipped
    dim3 ggrid(T_STEPS / 16, 48);
    k_gatesx<<<ggrid, 256, 0, stream>>>(inb, wxb, b0, gx);

    // persistent sequential scan
    k_scan<<<NB_SCAN, NT_SCAN, 0, stream>>>(gx, wef, w1b, wlb, wmb, b1, blin, bmap,
                                            embed0, h0, h1, out, syn);
}
